// LXformer_40303973105900
// MI455X (gfx1250) — matrix operand order repaired and checked by reading, not compiled, not run on hardware
//
#include <hip/hip_runtime.h>
#include <hip/hip_bf16.h>

typedef __attribute__((ext_vector_type(16))) _Float16 v16h;
typedef __attribute__((ext_vector_type(8)))  float    v8f;
typedef __attribute__((ext_vector_type(2)))  float    v2f;

#define BATCH 2
#define NPTS  8192
#define KNN   16
#define CH    32
#define ROWS  (BATCH * NPTS)     // 16384
#define HCNT  (ROWS * KNN)       // 262144 (q,k) pairs

// ---------------- workspace layout (float units) ----------------
// [0..15]  stats: sum[4], sumsq[4], mean[4], a[4]
// [64..)   idx : ROWS*16 ints
// then     lt  : ROWS*96 floats
// then     h   : ROWS*16*4 floats
#define WS_STATS 0
#define WS_IDX   64
#define WS_LT    (WS_IDX + ROWS * KNN)
#define WS_H     (WS_LT + ROWS * 96)

// =================================================================
// K0: zero the batchnorm accumulators (must be reset every launch)
// =================================================================
__global__ void k_zero_stats(float* stats) {
    if (threadIdx.x < 16) stats[threadIdx.x] = 0.0f;
}

// =================================================================
// K1: lt = features @ lt_w.T + lt_b   (16384x32 @ 32x96)
// One wave per 16-row tile; 6 col tiles; K=32 => one f16 WMMA per
// output tile (f32 accumulate).
// =================================================================
__global__ void k_lt_gemm(const float* __restrict__ feat,
                          const float* __restrict__ ltw,
                          const float* __restrict__ ltb,
                          float* __restrict__ lt) {
    const int wave = threadIdx.x >> 5;
    const int lane = threadIdx.x & 31;
    const int rt   = blockIdx.x * 8 + wave;     // 0..1023 row tiles
    const int g    = lane >> 4;                 // lane half-group
    const int nl   = lane & 15;

    // A tile: lane holds row m = rt*16 + nl
    // halves 0..7  -> K = g*8 + h ; halves 8..15 -> K = 16 + g*8 + (h-8)
    v16h a;
    const float* frow = feat + (rt * 16 + nl) * CH;
    #pragma unroll
    for (int h = 0; h < 16; ++h) {
        int k = (h < 8) ? (g * 8 + h) : (16 + g * 8 + (h - 8));
        a[h] = (_Float16)frow[k];
    }

    #pragma unroll
    for (int ct = 0; ct < 6; ++ct) {
        const int ncol = ct * 16 + nl;
        // B tile: lane holds col n = ncol; halves 0..7 -> K = g*8 + h ; halves 8..15 -> K = 16 + g*8 + (h-8)
        v16h b;
        #pragma unroll
        for (int h = 0; h < 16; ++h) {
            b[h] = (_Float16)ltw[ncol * CH + ((h < 8) ? (g * 8 + h) : (16 + g * 8 + (h - 8)))];   // lt_w is (96,32)
        }
        v8f c = {};
        c = __builtin_amdgcn_wmma_f32_16x16x32_f16(false, a, false, b,
                                                   (short)0, c, false, false);
        const float bias = ltb[ncol];
        #pragma unroll
        for (int v = 0; v < 8; ++v) {
            int m = rt * 16 + v + g * 8;
            lt[m * 96 + ncol] = c[v] + bias;
        }
    }
}

// =================================================================
// K2: KNN top-16 via full-f32 WMMA distance tiles.
// V_WMMA_F32_16X16X4_F32: A = 16 candidates x 4 coords (x,y,t,0),
// B = 4 x 16 queries. A/B are 2 VGPRs each:
//   lane L: pair base k0 = (L>>4)*2 ; A: row m = L&15, (a.x,a.y) =
//   coords[k0],coords[k0+1]; B: col n = L&15 likewise.
// D: lane L -> query n = L&15, candidate rows m = v + (L>>4)*8.
// Candidates staged in LDS chunks of 256 (+ prefetch of next chunk);
// per-lane sorted top-16 in registers; lane pairs merged by shfl_xor.
// =================================================================
__global__ void k_knn(const float* __restrict__ xytp, int* __restrict__ idxout) {
    __shared__ float spts[256 * 4];               // x,y,t,sq per candidate
    const int bi   = blockIdx.y;
    const int lane = threadIdx.x & 31;
    const int wave = threadIdx.x >> 5;
    const int g    = lane >> 4;
    const int nl   = lane & 15;
    const int qt   = blockIdx.x * 8 + wave;       // query tile 0..511
    const int q    = qt * 16 + nl;                // this lane's query
    const float* P = xytp + bi * NPTS * 4;

    const float qx = P[q * 4 + 0], qy = P[q * 4 + 1], qtc = P[q * 4 + 2];
    const float sqQ = qx * qx + qy * qy + qtc * qtc;

    // B (4x16): lanes 0-15 carry (K0,K1)=(x,y); lanes 16-31 carry (K2,K3)=(t,0)
    v2f bq;
    bq.x = (g == 0) ? qx : qtc;
    bq.y = (g == 0) ? qy : 0.0f;

    float dist[KNN]; int idxr[KNN];
    #pragma unroll
    for (int j = 0; j < KNN; ++j) { dist[j] = 3.4e38f; idxr[j] = 0; }

    for (int chunk = 0; chunk < NPTS; chunk += 256) {
        __syncthreads();
        {
            const int t = threadIdx.x;
            float x  = P[(chunk + t) * 4 + 0];
            float y  = P[(chunk + t) * 4 + 1];
            float tt = P[(chunk + t) * 4 + 2];
            spts[t * 4 + 0] = x; spts[t * 4 + 1] = y; spts[t * 4 + 2] = tt;
            spts[t * 4 + 3] = x * x + y * y + tt * tt;
            if (chunk + 256 < NPTS)   // overlap next chunk fetch (global_prefetch_b8)
                __builtin_prefetch(P + (chunk + 256 + t) * 4, 0, 1);
        }
        __syncthreads();

        for (int tb = 0; tb < 256; tb += 16) {
            // A (16x4): lane row m = tb + nl
            v2f am;
            am.x = spts[(tb + nl) * 4 + ((g == 0) ? 0 : 2)];
            am.y = (g == 0) ? spts[(tb + nl) * 4 + 1] : 0.0f;

            v8f dot = {};
            dot = __builtin_amdgcn_wmma_f32_16x16x4_f32(false, am, false, bq,
                                                        (short)0, dot, false, false);
            #pragma unroll
            for (int v = 0; v < 8; ++v) {
                const int mloc = tb + v + g * 8;
                const float d = spts[mloc * 4 + 3] + sqQ - 2.0f * dot[v];
                const int cid = chunk + mloc;
                if (d < dist[KNN - 1]) {            // register bubble-insert
                    dist[KNN - 1] = d; idxr[KNN - 1] = cid;
                    #pragma unroll
                    for (int j = KNN - 1; j > 0; --j) {
                        if (dist[j] < dist[j - 1]) {
                            float td = dist[j]; dist[j] = dist[j - 1]; dist[j - 1] = td;
                            int   ti = idxr[j]; idxr[j] = idxr[j - 1]; idxr[j - 1] = ti;
                        }
                    }
                }
            }
        }
    }

    // merge lane L with partner L^16 (same query, other 8-row slice)
    float pd[KNN]; int pi[KNN];
    #pragma unroll
    for (int j = 0; j < KNN; ++j) {
        pd[j] = __shfl_xor(dist[j], 16);
        pi[j] = __shfl_xor(idxr[j], 16);
    }
    #pragma unroll
    for (int j = 0; j < KNN; ++j) {
        if (pd[j] < dist[KNN - 1]) {
            dist[KNN - 1] = pd[j]; idxr[KNN - 1] = pi[j];
            #pragma unroll
            for (int t = KNN - 1; t > 0; --t) {
                if (dist[t] < dist[t - 1]) {
                    float td = dist[t]; dist[t] = dist[t - 1]; dist[t - 1] = td;
                    int   ti = idxr[t]; idxr[t] = idxr[t - 1]; idxr[t - 1] = ti;
                }
            }
        }
    }
    if (lane < 16) {
        int* out = idxout + (bi * NPTS + q) * KNN;
        #pragma unroll
        for (int j = 0; j < KNN; ++j) out[j] = idxr[j];
    }
}

// =================================================================
// K3: h = rel @ pe_w1.T + pe_b1 (store), + BatchNorm partial sums.
// =================================================================
__global__ void k_pe_h(const float* __restrict__ xytp, const int* __restrict__ idx,
                       const float* __restrict__ w1, const float* __restrict__ b1,
                       float* __restrict__ h, float* __restrict__ stats) {
    const int tid = blockIdx.x * 256 + threadIdx.x;   // 0..HCNT-1
    const int row = tid >> 4;
    const int k   = tid & 15;
    const int bi  = row >> 13;
    const int nb  = idx[row * KNN + k];
    const float* pq = xytp + row * 4;
    const float* pn = xytp + (bi * NPTS + nb) * 4;
    const float r0 = pq[0] - pn[0], r1 = pq[1] - pn[1];
    const float r2 = pq[2] - pn[2], r3 = pq[3] - pn[3];

    float hv[4];
    #pragma unroll
    for (int f = 0; f < 4; ++f) {
        hv[f] = b1[f] + r0 * w1[f * 4 + 0] + r1 * w1[f * 4 + 1]
                      + r2 * w1[f * 4 + 2] + r3 * w1[f * 4 + 3];
        h[tid * 4 + f] = hv[f];
    }
    const int lane = threadIdx.x & 31;
    #pragma unroll
    for (int f = 0; f < 4; ++f) {
        float s = hv[f], s2 = hv[f] * hv[f];
        #pragma unroll
        for (int off = 16; off >= 1; off >>= 1) {
            s  += __shfl_xor(s,  off);
            s2 += __shfl_xor(s2, off);
        }
        if (lane == 0) {
            atomicAdd(&stats[f],     s);
            atomicAdd(&stats[4 + f], s2);
        }
    }
}

// =================================================================
// K4: finalize BN stats: mean[4], a[4] = rsqrt(var+eps)*gamma
// =================================================================
__global__ void k_finalize(const float* __restrict__ gamma, float* __restrict__ stats) {
    if (threadIdx.x < 4) {
        const float inv_cnt = 1.0f / (float)HCNT;
        const float mean = stats[threadIdx.x] * inv_cnt;
        const float var  = stats[4 + threadIdx.x] * inv_cnt - mean * mean;
        stats[8  + threadIdx.x] = mean;
        stats[12 + threadIdx.x] = rsqrtf(var + 1e-5f) * gamma[threadIdx.x];
    }
}

// =================================================================
// K5: fused epilogue. One wave per point; lane = channel.
// =================================================================
__global__ void k_attn(const float* __restrict__ lt, const int* __restrict__ idx,
                       const float* __restrict__ h, const float* __restrict__ stats,
                       const float* __restrict__ pe_beta, const float* __restrict__ w2,
                       const float* __restrict__ b2, const float* __restrict__ ln_g,
                       const float* __restrict__ ln_b, float* __restrict__ out) {
    const int wave = threadIdx.x >> 5;
    const int lane = threadIdx.x & 31;
    const int row  = blockIdx.x * 8 + wave;        // 0..ROWS-1
    const int bi   = row >> 13;
    const int base = bi * NPTS;
    const int c    = lane;

    const float w20 = w2[c * 4 + 0], w21 = w2[c * 4 + 1];
    const float w22 = w2[c * 4 + 2], w23 = w2[c * 4 + 3];
    const float bias2 = b2[c];
    const float m0 = stats[8],  m1 = stats[9],  m2 = stats[10], m3 = stats[11];
    const float a0 = stats[12], a1 = stats[13], a2 = stats[14], a3 = stats[15];
    const float be0 = pe_beta[0], be1 = pe_beta[1], be2 = pe_beta[2], be3 = pe_beta[3];
    const float varphi = lt[row * 96 + c];
    const float gln = ln_g[c], bln = ln_b[c];
    const float inv_scale = 0.17677669529663687f;  // 1/sqrt(32)

    float s[KNN], vv[KNN];
    #pragma unroll
    for (int k = 0; k < KNN; ++k) {
        const float* hh = h + (row * KNN + k) * 4;
        const float rl0 = fmaxf(0.f, (hh[0] - m0) * a0 + be0);
        const float rl1 = fmaxf(0.f, (hh[1] - m1) * a1 + be1);
        const float rl2 = fmaxf(0.f, (hh[2] - m2) * a2 + be2);
        const float rl3 = fmaxf(0.f, (hh[3] - m3) * a3 + be3);
        const float delta = bias2 + rl0 * w20 + rl1 * w21 + rl2 * w22 + rl3 * w23;

        const int nb = idx[row * KNN + k];
        const float* ltn = lt + (base + nb) * 96;
        const float pre = varphi - ltn[CH + c] + delta;

        float sum = pre, sum2 = pre * pre;
        #pragma unroll
        for (int off = 16; off >= 1; off >>= 1) {
            sum  += __shfl_xor(sum,  off);
            sum2 += __shfl_xor(sum2, off);
        }
        const float mu  = sum * (1.0f / 32.0f);
        const float var = sum2 * (1.0f / 32.0f) - mu * mu;
        const float ln  = (pre - mu) * rsqrtf(var + 1e-5f) * gln + bln;
        s[k]  = ln * inv_scale;
        vv[k] = ltn[2 * CH + c] + delta;
    }

    float mx = s[0];
    #pragma unroll
    for (int k = 1; k < KNN; ++k) mx = fmaxf(mx, s[k]);
    float den = 0.f, acc = 0.f;
    #pragma unroll
    for (int k = 0; k < KNN; ++k) {
        const float e = __expf(s[k] - mx);
        den += e;
        acc += e * vv[k];
    }
    out[row * CH + c] = acc / den;
}

// =================================================================
extern "C" void kernel_launch(void* const* d_in, const int* in_sizes, int n_in,
                              void* d_out, int out_size, void* d_ws, size_t ws_size,
                              hipStream_t stream) {
    const float* xytp     = (const float*)d_in[0];
    const float* features = (const float*)d_in[1];
    const float* pe_w1    = (const float*)d_in[2];
    const float* pe_b1    = (const float*)d_in[3];
    const float* pe_gamma = (const float*)d_in[4];
    const float* pe_beta  = (const float*)d_in[5];
    const float* pe_w2    = (const float*)d_in[6];
    const float* pe_b2    = (const float*)d_in[7];
    const float* lt_w     = (const float*)d_in[8];
    const float* lt_b     = (const float*)d_in[9];
    const float* ln_gamma = (const float*)d_in[10];
    const float* ln_beta  = (const float*)d_in[11];
    float* out = (float*)d_out;

    float* ws    = (float*)d_ws;
    float* stats = ws + WS_STATS;
    int*   idx   = (int*)(ws + WS_IDX);
    float* lt    = ws + WS_LT;
    float* h     = ws + WS_H;

    k_zero_stats<<<1, 32, 0, stream>>>(stats);
    k_lt_gemm<<<128, 256, 0, stream>>>(features, lt_w, lt_b, lt);
    k_knn<<<dim3(64, BATCH), 256, 0, stream>>>(xytp, idx);
    k_pe_h<<<HCNT / 256, 256, 0, stream>>>(xytp, idx, pe_w1, pe_b1, h, stats);
    k_finalize<<<1, 32, 0, stream>>>(pe_gamma, stats);
    k_attn<<<ROWS / 8, 256, 0, stream>>>(lt, idx, h, stats, pe_beta, pe_w2, pe_b2,
                                         ln_gamma, ln_beta, out);
}
